// RotarySelfAttention_80324478370015
// MI455X (gfx1250) — matrix operand order repaired and checked by reading, not compiled, not run on hardware
//
#include <hip/hip_runtime.h>
#include <hip/hip_bf16.h>

// Problem constants (B, N, DIM, H, DH) = (2, 2048, 1024, 16, 64)
#define BB     2
#define NN     2048
#define DIMM   1024
#define HH     16
#define DHH    64
#define INNERR 1024
#define ROWS   (BB * NN)      // 4096
#define QKVC   (3 * INNERR)   // 3072
#define SCALE  0.125f         // DH^-0.5

typedef __bf16 bf16_t;
typedef __attribute__((ext_vector_type(16))) __bf16 v16bf;
typedef __attribute__((ext_vector_type(8)))  float  v8f;

// ---------------------------------------------------------------------------
// CDNA5 async global->LDS copy (ASYNCcnt path, CDNA5 ISA)
// LDS byte offset = low 32 bits of the flat LDS address (ISA 10.2 aperture).
// ---------------------------------------------------------------------------
__device__ __forceinline__ void async_copy_b128(void* lds_dst,
                                                const void* gsrc) {
  unsigned loff = (unsigned)(size_t)lds_dst;
  asm volatile("global_load_async_to_lds_b128 %0, %1, off"
               :
               : "v"(loff), "v"(gsrc)
               : "memory");
}
__device__ __forceinline__ void async_wait0() {
  asm volatile("s_wait_asynccnt 0x0" ::: "memory");
}

// ---------------------------------------------------------------------------
// WMMA helpers (wave32, 16x16x32 bf16 -> f32)
// ---------------------------------------------------------------------------
__device__ __forceinline__ v8f wmma_bf16(v16bf a, v16bf b, v8f c) {
  return __builtin_amdgcn_wmma_f32_16x16x32_bf16(false, a, false, b,
                                                 (short)0, c, false, false);
}

// A-matrix 16x32 bf16 fragment from LDS, row-major source [rows][ld].
__device__ __forceinline__ v16bf load_a_frag(const bf16_t* base, int ld,
                                             int row0, int kbase, int lane) {
  const int m = row0 + (lane & 15);
  const int half = lane >> 4;
  const unsigned int* rp =
      reinterpret_cast<const unsigned int*>(base + (size_t)m * ld + kbase);
  v16bf a;
  unsigned int* ap = reinterpret_cast<unsigned int*>(&a);
#pragma unroll
  for (int v = 0; v < 8; ++v) {
    const int k0 = ((v & 4) << 2) + ((v & 3) << 1) + (half << 3);
    ap[v] = rp[k0 >> 1];
  }
  return a;
}

// B-matrix 32x16 bf16 fragment; operand staged column(n)-major: [n][ld].
__device__ __forceinline__ v16bf load_b_frag(const bf16_t* base, int ld,
                                             int col0, int kbase, int lane) {
  const int n = col0 + (lane & 15);
  const int half = lane >> 4;
  const unsigned int* cp =
      reinterpret_cast<const unsigned int*>(base + (size_t)n * ld + kbase);
  v16bf b;
  unsigned int* bp = reinterpret_cast<unsigned int*>(&b);
#pragma unroll
  for (int v = 0; v < 8; ++v) {
    const int k0 = ((v & 4) << 2) + ((v & 3) << 1) + (half << 3);
    bp[v] = cp[k0 >> 1];
  }
  return b;
}

// ---------------------------------------------------------------------------
// Tiled transpose + f32 -> bf16 convert: dst[n][k] = (bf16) src[k][n]
// Block = 256 threads handling a 32x32 tile through LDS.
// ---------------------------------------------------------------------------
__global__ __launch_bounds__(256)
void transpose_bf16_kernel(const float* __restrict__ src,  // [K][N]
                           bf16_t* __restrict__ dst,       // [N][K]
                           int K, int N) {
  __shared__ float tile[32][33];
  const int k0 = blockIdx.y * 32, n0 = blockIdx.x * 32;
  const int tx = threadIdx.x & 31, ty = threadIdx.x >> 5;  // 32 x 8
#pragma unroll
  for (int i = 0; i < 32; i += 8)
    tile[ty + i][tx] = src[(size_t)(k0 + ty + i) * N + n0 + tx];
  __syncthreads();
#pragma unroll
  for (int i = 0; i < 32; i += 8)
    dst[(size_t)(n0 + ty + i) * K + k0 + tx] = (bf16_t)tile[tx][ty + i];
}

// ---------------------------------------------------------------------------
// LayerNorm over DIM=1024, one block (8 waves) per row, writes bf16 h
// ---------------------------------------------------------------------------
__global__ __launch_bounds__(256)
void layernorm_kernel(const float* __restrict__ x, const float* __restrict__ g,
                      const float* __restrict__ bta, bf16_t* __restrict__ h) {
  const int row = blockIdx.x;
  const int t = threadIdx.x;
  const float* xr = x + (size_t)row * DIMM;
  float v[4];
  float s = 0.f, s2 = 0.f;
#pragma unroll
  for (int i = 0; i < 4; ++i) {
    v[i] = xr[t + 256 * i];
    s += v[i];
    s2 += v[i] * v[i];
  }
#pragma unroll
  for (int off = 16; off; off >>= 1) {
    s  += __shfl_xor(s, off, 32);
    s2 += __shfl_xor(s2, off, 32);
  }
  __shared__ float rs[8], rs2[8];
  const int wv = t >> 5, lane = t & 31;
  if (lane == 0) { rs[wv] = s; rs2[wv] = s2; }
  __syncthreads();
  if (wv == 0) {
    float a  = (lane < 8) ? rs[lane]  : 0.f;
    float a2 = (lane < 8) ? rs2[lane] : 0.f;
#pragma unroll
    for (int off = 4; off; off >>= 1) {
      a  += __shfl_xor(a, off, 32);
      a2 += __shfl_xor(a2, off, 32);
    }
    if (lane == 0) { rs[0] = a; rs2[0] = a2; }
  }
  __syncthreads();
  const float mu = rs[0] * (1.f / DIMM);
  const float var = rs2[0] * (1.f / DIMM) - mu * mu;
  const float inv = rsqrtf(var + 1e-5f);
#pragma unroll
  for (int i = 0; i < 4; ++i) {
    const int c = t + 256 * i;
    h[(size_t)row * DIMM + c] = (bf16_t)((v[i] - mu) * inv * g[c] + bta[c]);
  }
}

// ---------------------------------------------------------------------------
// QKV GEMM (4096x1024 x 1024x3072) + bias + rotary, scatter to head layout.
// 256 threads = 8 waves; block tile 128x128; wave 64x32; K-step 32;
// double-buffered LDS; ALL staging is async global->LDS B128 copies
// (weights pre-transposed to n-major). V written in (B,H,DH,N) layout.
// ---------------------------------------------------------------------------
__global__ __launch_bounds__(256)
void qkv_gemm_rotary_kernel(const bf16_t* __restrict__ A,    // [ROWS][DIM]
                            const bf16_t* __restrict__ BwT,  // [QKVC][DIM]
                            const float* __restrict__ bias,  // [QKVC]
                            const float* __restrict__ rotary,// [B][N][DH]
                            bf16_t* __restrict__ qh, bf16_t* __restrict__ kh,
                            bf16_t* __restrict__ vT) {
  __shared__ __align__(16) bf16_t ldsA[2][128 * 32];  // row-major [m][k]
  __shared__ __align__(16) bf16_t ldsB[2][128 * 32];  // n-major   [n][k]
  const int tid = threadIdx.x, lane = tid & 31, wv = tid >> 5;
  const int wr = wv >> 2, wc = wv & 3;
  const int rowBlock = blockIdx.y * 128, colBlock = blockIdx.x * 128;

  v8f acc[4][2] = {};

  // Both tiles: 128 rows x 32 cols = 512 x 16B linear chunks, async copied.
  auto stageA = [&](int k0, int buf) {
#pragma unroll
    for (int i = 0; i < 2; ++i) {
      const int c = tid + 256 * i;
      const int r = c >> 2, q = c & 3;
      async_copy_b128(&ldsA[buf][r * 32 + q * 8],
                      A + (size_t)(rowBlock + r) * DIMM + k0 + q * 8);
    }
  };
  auto stageB = [&](int k0, int buf) {
#pragma unroll
    for (int i = 0; i < 2; ++i) {
      const int c = tid + 256 * i;
      const int r = c >> 2, q = c & 3;
      async_copy_b128(&ldsB[buf][r * 32 + q * 8],
                      BwT + (size_t)(colBlock + r) * DIMM + k0 + q * 8);
    }
  };

  stageA(0, 0);
  stageB(0, 0);
  async_wait0();
  __syncthreads();

  for (int k0 = 0; k0 < DIMM; k0 += 32) {
    const int buf = (k0 >> 5) & 1;
    if (k0 + 32 < DIMM) {  // prefetch next K-step into the other buffer
      stageA(k0 + 32, buf ^ 1);
      stageB(k0 + 32, buf ^ 1);
    }
    v16bf af[4];
#pragma unroll
    for (int i = 0; i < 4; ++i)
      af[i] = load_a_frag(ldsA[buf], 32, wr * 64 + i * 16, 0, lane);
#pragma unroll
    for (int j = 0; j < 2; ++j) {
      v16bf bfr = load_b_frag(ldsB[buf], 32, wc * 32 + j * 16, 0, lane);
#pragma unroll
      for (int i = 0; i < 4; ++i) acc[i][j] = wmma_bf16(af[i], bfr, acc[i][j]);
    }
    async_wait0();
    __syncthreads();
  }

  // Epilogue: bias + rotary (partner column = adjacent lane) + head scatter
  const int n = lane & 15, half = lane >> 4;
#pragma unroll
  for (int i = 0; i < 4; ++i) {
#pragma unroll
    for (int j = 0; j < 2; ++j) {
      const int colT = colBlock + wc * 32 + j * 16;
      const int rowT = rowBlock + wr * 64 + i * 16;
      const int c = colT + n;
      const int mtx = c >> 10, rem = c & 1023;
      const int hd = rem >> 6, d = rem & 63;
#pragma unroll
      for (int e = 0; e < 8; ++e) {
        const int r = rowT + e + 8 * half;
        const int bidx = r >> 11, nseq = r & (NN - 1);
        const float t = acc[i][j][e] + bias[c];
        const float p = __shfl_xor(t, 1, 32);  // rotate-half partner
        const float rot = (d & 1) ? p : -p;
        const float f = rotary[((size_t)bidx * NN + nseq) * DHH + d];
        const float val = t * __cosf(f) + rot * __sinf(f);
        const bf16_t ov = (bf16_t)val;
        if (mtx == 2) {  // V stored dh-major: [(b,h,d)][n]
          vT[((size_t)(bidx * HH + hd) * DHH + d) * NN + nseq] = ov;
        } else {
          bf16_t* dst = (mtx == 0) ? qh : kh;
          dst[((size_t)(bidx * HH + hd) * NN + nseq) * DHH + d] = ov;
        }
      }
    }
  }
}

// ---------------------------------------------------------------------------
// Flash attention per (b,h): 128 thr = 4 waves, 64 query rows per block,
// keys streamed in chunks of 32 with double-buffered async K/V staging.
// Ends with inverse rotary.
// ---------------------------------------------------------------------------
__global__ __launch_bounds__(128)
void attention_kernel(const bf16_t* __restrict__ qh,
                      const bf16_t* __restrict__ kh,
                      const bf16_t* __restrict__ vT,  // [(b,h,d)][N]
                      const float* __restrict__ rotary,
                      bf16_t* __restrict__ aout) {
  __shared__ __align__(16) bf16_t ldsQ[64 * 64];        // [row][dh]
  __shared__ __align__(16) bf16_t ldsK[2][32 * 64];     // [key][dh]
  __shared__ __align__(16) bf16_t ldsV[2][64 * 32];     // [dh][key]
  __shared__ __align__(16) bf16_t ldsP[4][16 * 32];     // per-wave P tile

  const int tid = threadIdx.x, lane = tid & 31, wv = tid >> 5;
  const int bh = blockIdx.y;
  const int b = bh / HH, h = bh % HH;
  const int q0 = blockIdx.x * 64;
  const size_t headBase = (size_t)bh * NN * DHH;
  const int myRow0 = wv * 16;

  // K chunk: 4KB linear; V chunk: 64 dh-rows x 32 keys (rows linear in vT)
  auto stageKV = [&](int kc, int buf) {
    const bf16_t* ksrc = kh + headBase + (size_t)kc * DHH;
    const bf16_t* vsrc = vT + headBase + kc;
#pragma unroll
    for (int i = 0; i < 2; ++i) {
      const int c = tid + 128 * i;
      async_copy_b128(&ldsK[buf][c * 8], ksrc + c * 8);
      const int d = c >> 2, q = c & 3;
      async_copy_b128(&ldsV[buf][d * 32 + q * 8],
                      vsrc + (size_t)d * NN + q * 8);
    }
  };

  // Stage Q block once (8KB linear) + first K/V chunk
  {
    const bf16_t* src = qh + headBase + (size_t)q0 * DHH;
#pragma unroll
    for (int i = 0; i < 4; ++i) {
      const int c = tid + 128 * i;
      async_copy_b128(&ldsQ[c * 8], src + c * 8);
    }
  }
  stageKV(0, 0);
  async_wait0();
  __syncthreads();

  float m_i[8], l_i[8];
  v8f o[4] = {};
#pragma unroll
  for (int e = 0; e < 8; ++e) { m_i[e] = -1e30f; l_i[e] = 0.f; }

  for (int kc = 0; kc < NN; kc += 32) {
    const int buf = (kc >> 5) & 1;
    if (kc + 32 < NN) stageKV(kc + 32, buf ^ 1);  // async prefetch

    // S = Q K^T  (16 rows x 32 keys, two 16x16 C tiles, K-dim = 64 = 2x32)
    v8f s0 = {}, s1 = {};
#pragma unroll
    for (int dk = 0; dk < 64; dk += 32) {
      v16bf aq  = load_a_frag(ldsQ, 64, myRow0, dk, lane);
      v16bf bk0 = load_b_frag(ldsK[buf], 64, 0,  dk, lane);
      v16bf bk1 = load_b_frag(ldsK[buf], 64, 16, dk, lane);
      s0 = wmma_bf16(aq, bk0, s0);
      s1 = wmma_bf16(aq, bk1, s1);
    }

    // Online softmax (row r = e + 8*half lives in one 16-lane half-group)
    float p0[8], p1[8];
#pragma unroll
    for (int e = 0; e < 8; ++e) {
      const float a0 = s0[e] * SCALE, a1 = s1[e] * SCALE;
      float mx = fmaxf(a0, a1);
#pragma unroll
      for (int off = 8; off; off >>= 1) mx = fmaxf(mx, __shfl_xor(mx, off, 32));
      const float mnew = fmaxf(m_i[e], mx);
      const float alpha = __expf(m_i[e] - mnew);
      const float e0 = __expf(a0 - mnew), e1 = __expf(a1 - mnew);
      float rsum = e0 + e1;
#pragma unroll
      for (int off = 8; off; off >>= 1) rsum += __shfl_xor(rsum, off, 32);
      l_i[e] = l_i[e] * alpha + rsum;
      m_i[e] = mnew;
      p0[e] = e0; p1[e] = e1;
#pragma unroll
      for (int j = 0; j < 4; ++j) o[j][e] *= alpha;
    }

    // Write P (bf16) row-major [16][32], reload as A fragment
    {
      bf16_t* P = ldsP[wv];
      const int n = lane & 15, half = lane >> 4;
#pragma unroll
      for (int e = 0; e < 8; ++e) {
        const int m = e + 8 * half;
        P[m * 32 + n]      = (bf16_t)p0[e];
        P[m * 32 + n + 16] = (bf16_t)p1[e];
      }
    }
    __syncthreads();

    // O += P V   (B fragments from dh-major ldsV)
    {
      v16bf ap = load_a_frag(ldsP[wv], 32, 0, 0, lane);
#pragma unroll
      for (int j = 0; j < 4; ++j) {
        v16bf bv = load_b_frag(ldsV[buf], 32, j * 16, 0, lane);
        o[j] = wmma_bf16(ap, bv, o[j]);
      }
    }
    async_wait0();
    __syncthreads();
  }

  // Finalize: 1/l, inverse rotary (cos f, -sin f), write bf16 (B,N,INNER)
  const int n = lane & 15, half = lane >> 4;
#pragma unroll
  for (int e = 0; e < 8; ++e) {
    const int row = q0 + myRow0 + e + 8 * half;
    const float inv = 1.f / l_i[e];
#pragma unroll
    for (int j = 0; j < 4; ++j) {
      const int d = j * 16 + n;
      const float val = o[j][e] * inv;
      const float p = __shfl_xor(val, 1, 32);
      const float rot = (d & 1) ? p : -p;
      const float f = rotary[((size_t)b * NN + row) * DHH + d];
      const float res = val * __cosf(f) - rot * __sinf(f);
      aout[((size_t)b * NN + row) * INNERR + h * DHH + d] = (bf16_t)res;
    }
  }
}

// ---------------------------------------------------------------------------
// Out GEMM (4096x1024 x 1024x1024) + bias -> f32 output; double-buffered,
// all-async staging (weights pre-transposed to n-major).
// ---------------------------------------------------------------------------
__global__ __launch_bounds__(256)
void out_gemm_kernel(const bf16_t* __restrict__ A,    // [ROWS][INNER]
                     const bf16_t* __restrict__ BwT,  // [DIM][INNER] n-major
                     const float* __restrict__ bias,  // [DIM]
                     float* __restrict__ out) {
  __shared__ __align__(16) bf16_t ldsA[2][128 * 32];
  __shared__ __align__(16) bf16_t ldsB[2][128 * 32];
  const int tid = threadIdx.x, lane = tid & 31, wv = tid >> 5;
  const int wr = wv >> 2, wc = wv & 3;
  const int rowBlock = blockIdx.y * 128, colBlock = blockIdx.x * 128;

  v8f acc[4][2] = {};

  auto stageA = [&](int k0, int buf) {
#pragma unroll
    for (int i = 0; i < 2; ++i) {
      const int c = tid + 256 * i;
      const int r = c >> 2, q = c & 3;
      async_copy_b128(&ldsA[buf][r * 32 + q * 8],
                      A + (size_t)(rowBlock + r) * INNERR + k0 + q * 8);
    }
  };
  auto stageB = [&](int k0, int buf) {
#pragma unroll
    for (int i = 0; i < 2; ++i) {
      const int c = tid + 256 * i;
      const int r = c >> 2, q = c & 3;
      async_copy_b128(&ldsB[buf][r * 32 + q * 8],
                      BwT + (size_t)(colBlock + r) * INNERR + k0 + q * 8);
    }
  };

  stageA(0, 0);
  stageB(0, 0);
  async_wait0();
  __syncthreads();

  for (int k0 = 0; k0 < INNERR; k0 += 32) {
    const int buf = (k0 >> 5) & 1;
    if (k0 + 32 < INNERR) {
      stageA(k0 + 32, buf ^ 1);
      stageB(k0 + 32, buf ^ 1);
    }
    v16bf af[4];
#pragma unroll
    for (int i = 0; i < 4; ++i)
      af[i] = load_a_frag(ldsA[buf], 32, wr * 64 + i * 16, 0, lane);
#pragma unroll
    for (int j = 0; j < 2; ++j) {
      v16bf bfr = load_b_frag(ldsB[buf], 32, wc * 32 + j * 16, 0, lane);
#pragma unroll
      for (int i = 0; i < 4; ++i) acc[i][j] = wmma_bf16(af[i], bfr, acc[i][j]);
    }
    async_wait0();
    __syncthreads();
  }

  const int n = lane & 15, half = lane >> 4;
#pragma unroll
  for (int i = 0; i < 4; ++i) {
#pragma unroll
    for (int j = 0; j < 2; ++j) {
      const int c = colBlock + wc * 32 + j * 16 + n;
      const int rowT = rowBlock + wr * 64 + i * 16;
      const float bv = bias[c];
#pragma unroll
      for (int e = 0; e < 8; ++e) {
        const int r = rowT + e + 8 * half;
        out[(size_t)r * DIMM + c] = acc[i][j][e] + bv;
      }
    }
  }
}

// ---------------------------------------------------------------------------
// Launch
// ---------------------------------------------------------------------------
extern "C" void kernel_launch(void* const* d_in, const int* in_sizes, int n_in,
                              void* d_out, int out_size, void* d_ws,
                              size_t ws_size, hipStream_t stream) {
  (void)in_sizes; (void)n_in; (void)out_size; (void)ws_size;
  const float* x      = (const float*)d_in[0];
  const float* rotary = (const float*)d_in[1];
  const float* ln_g   = (const float*)d_in[2];
  const float* ln_b   = (const float*)d_in[3];
  const float* w_qkv  = (const float*)d_in[4];
  const float* b_qkv  = (const float*)d_in[5];
  const float* w_out  = (const float*)d_in[6];
  const float* b_out  = (const float*)d_in[7];
  float* out = (float*)d_out;

  char* ws = (char*)d_ws;
  size_t off = 0;
  auto alloc = [&](size_t bytes) -> void* {
    void* p = ws + off;
    off += (bytes + 255) & ~(size_t)255;
    return p;
  };
  bf16_t* hbuf   = (bf16_t*)alloc((size_t)ROWS * DIMM * 2);
  bf16_t* wqkvT  = (bf16_t*)alloc((size_t)DIMM * QKVC * 2);   // [QKVC][DIM]
  bf16_t* woutT  = (bf16_t*)alloc((size_t)INNERR * DIMM * 2); // [DIM][INNER]
  bf16_t* qhb    = (bf16_t*)alloc((size_t)BB * HH * NN * DHH * 2);
  bf16_t* khb    = (bf16_t*)alloc((size_t)BB * HH * NN * DHH * 2);
  bf16_t* vTb    = (bf16_t*)alloc((size_t)BB * HH * NN * DHH * 2);
  bf16_t* aoutb  = (bf16_t*)alloc((size_t)ROWS * INNERR * 2);

  // Transpose-convert weights: wqkvT[n][k] = w_qkv[k][n], etc.
  transpose_bf16_kernel<<<dim3(QKVC / 32, DIMM / 32), 256, 0, stream>>>(
      w_qkv, wqkvT, DIMM, QKVC);
  transpose_bf16_kernel<<<dim3(DIMM / 32, INNERR / 32), 256, 0, stream>>>(
      w_out, woutT, INNERR, DIMM);
  layernorm_kernel<<<ROWS, 256, 0, stream>>>(x, ln_g, ln_b, hbuf);
  qkv_gemm_rotary_kernel<<<dim3(QKVC / 128, ROWS / 128), 256, 0, stream>>>(
      hbuf, wqkvT, b_qkv, rotary, qhb, khb, vTb);
  attention_kernel<<<dim3(NN / 64, BB * HH), 128, 0, stream>>>(
      qhb, khb, vTb, rotary, aoutb);
  out_gemm_kernel<<<dim3(DIMM / 128, ROWS / 128), 256, 0, stream>>>(
      aoutb, woutT, b_out, out);
}
